// MultiFrequencyFeatureExtraction_21320217658115
// MI455X (gfx1250) — matrix operand order repaired and checked by reading, not compiled, not run on hardware
//
#include <hip/hip_runtime.h>
#include <hip/hip_bf16.h>

// ---------------------------------------------------------------------------
// MultiFrequencyFeatureExtraction for MI455X (gfx1250, wave32)
//   Kernel 1: fused x*f_k pooling. Filters staged into 200KB LDS via the
//             CDNA5 Tensor Data Mover (tensor_load_to_lds + s_wait_tensorcnt),
//             4 channels per wave + float2 per lane -> VALU-bound loop with
//             ds_load_b64 / global_load_b64 data movement.
//   Kernel 2: tiny MLP via v_wmma_f32_16x16x32_f16:
//             Hsum = sum_v relu(V @ w1^T), out = Hsum @ w2^T.
// ---------------------------------------------------------------------------

#define HW   3136   // 56*56
#define KF   16     // frequency branches
#define CCH  512    // channels
#define BB   32     // batch
#define MID  32     // C/16
#define CPW  4      // channels per wave in kernel 1

typedef __attribute__((ext_vector_type(16))) _Float16 v16h;
typedef __attribute__((ext_vector_type(8)))  float    v8f;

typedef __attribute__((ext_vector_type(4))) unsigned int tdm_u32x4;
typedef __attribute__((ext_vector_type(8))) int          tdm_i32x8;
typedef __attribute__((ext_vector_type(4))) int          tdm_i32x4;

// ---------------------------------------------------------------------------
// Kernel 1: per-(b,c) pooled stats.
// grid = 512 blocks x 256 threads (8 waves). Each wave owns 4 consecutive
// channels of one batch image; lanes partition the 3136 HW positions in
// float2 chunks (3136 = 49 * 64).
// ---------------------------------------------------------------------------
__global__ __launch_bounds__(256) void freq_pool_kernel(
    const float* __restrict__ x,    // [B, C, H, W]
    const float* __restrict__ dct,  // [K, H, W]
    float* __restrict__ sa,         // [B*C]
    float* __restrict__ sm,         // [B*C]
    float* __restrict__ sn)         // [B*C]
{
    __shared__ float sF[KF * HW];   // 200704 bytes of LDS (fits 320KB WGP)

    const int tid = threadIdx.x;

#if __has_builtin(__builtin_amdgcn_tensor_load_to_lds)
    // ---- CDNA5 TDM: one async DMA of the whole [K,H,W] filter into LDS ----
    if (tid < 32) {                 // wave 0 issues (TDM ignores EXEC)
        const unsigned lds_off = (unsigned)(uintptr_t)(void*)&sF[0];
        const unsigned long long ga = (unsigned long long)(uintptr_t)dct;

        // D# group0: count=1 | lds_addr | global_addr[56:0] | type=2
        const tdm_u32x4 g0 = {
            1u,                                        // count=1 (valid), flags=0
            lds_off,                                   // bits 63:32  lds_addr
            (unsigned)(ga & 0xFFFFFFFFu),              // bits 95:64  global_addr lo
            (unsigned)((ga >> 32) & 0x01FFFFFFu) |     // bits 120:96 global_addr hi
                (2u << 30)                             // bits 127:126 type=2
        };
        // D# group1: 1-row tensor, dim0 = 50176 elements of 4 bytes
        const tdm_i32x8 g1 = {
            (int)(2u << 16),                 // workgroup_mask=0, data_size=2 (4B)
            (int)((KF * HW & 0xFFFF) << 16), // tensor_dim0[15:0]  -> bits 63:48
            (int)(((KF * HW) >> 16) |        // tensor_dim0[31:16] -> bits 79:64
                  (1 << 16)),                // tensor_dim1=1      -> bits 95:80
            (int)((KF * HW & 0xFFFF) << 16), // tile_dim0=50176    -> bits 127:112
            1,                               // tile_dim1=1, tile_dim2=0
            KF * HW,                         // tensor_dim0_stride lo32
            0,                               // tensor_dim0_stride hi16, dim1_stride lo
            0
        };
        const tdm_i32x4 gz = {0, 0, 0, 0};   // groups 2/3 unused (<=2D tensor)
#if __clang_major__ >= 23
        const tdm_i32x8 gz8 = {0, 0, 0, 0, 0, 0, 0, 0};
        __builtin_amdgcn_tensor_load_to_lds(g0, g1, gz, gz, gz8, 0);
#else
        __builtin_amdgcn_tensor_load_to_lds(g0, g1, gz, gz, 0);
#endif
        __builtin_amdgcn_s_wait_tensorcnt(0);
    }
#else
    for (int j = tid; j < KF * HW; j += 256) sF[j] = dct[j];
#endif
    __syncthreads();

    const int wave = tid >> 5;
    const int lane = tid & 31;
    const int b     = blockIdx.x >> 4;                       // 0..31
    const int cbase = ((blockIdx.x & 15) << 5) + wave * CPW; // 0..508
    const float* xp = x + ((size_t)b * CCH + cbase) * HW;

    float s[CPW];
    float mx[CPW][KF];
    float mn[CPW][KF];
#pragma unroll
    for (int cc = 0; cc < CPW; ++cc) {
        s[cc] = 0.0f;
#pragma unroll
        for (int k = 0; k < KF; ++k) {
            mx[cc][k] = -__builtin_inff();
            mn[cc][k] =  __builtin_inff();
        }
    }

    for (int i = 0; i < HW / 64; ++i) {          // 49 iterations, no remainder
        const int hw = (lane << 1) + (i << 6);   // float2 per lane, coalesced
        float2 xv[CPW];
#pragma unroll
        for (int cc = 0; cc < CPW; ++cc)
            xv[cc] = *(const float2*)(xp + cc * HW + hw);    // global_load_b64
#pragma unroll
        for (int k = 0; k < KF; ++k) {
            const float2 f = *(const float2*)(&sF[k * HW + hw]); // ds_load_b64
#pragma unroll
            for (int cc = 0; cc < CPW; ++cc) {
                const float xf0 = xv[cc].x * f.x;
                const float xf1 = xv[cc].y * f.y;
                s[cc] += xf0;
                s[cc] += xf1;
                mx[cc][k] = fmaxf(mx[cc][k], fmaxf(xf0, xf1));
                mn[cc][k] = fminf(mn[cc][k], fminf(xf0, xf1));
            }
        }
    }

    // wave32 reductions (xor tree across lanes)
#pragma unroll
    for (int cc = 0; cc < CPW; ++cc) {
        float ssum = s[cc];
        for (int off = 16; off > 0; off >>= 1) ssum += __shfl_xor(ssum, off, 32);

        float smax = 0.0f, smin = 0.0f;
#pragma unroll
        for (int k = 0; k < KF; ++k) {
            float m = mx[cc][k];
            for (int off = 16; off > 0; off >>= 1) m = fmaxf(m, __shfl_xor(m, off, 32));
            smax += m;
            float n = mn[cc][k];
            for (int off = 16; off > 0; off >>= 1) n = fminf(n, __shfl_xor(n, off, 32));
            smin += n;
        }

        if (lane == 0) {
            const int bc = b * CCH + cbase + cc;
            sa[bc] = ssum * (1.0f / (float)(HW * KF));
            sm[bc] = smax * (1.0f / (float)KF);
            sn[bc] = smin * (1.0f / (float)KF);
        }
    }
}

// ---------------------------------------------------------------------------
// Kernel 2: MLP with WMMA. One workgroup of 256 threads (8 waves).
// WMMA 16x16x32 f16 operand layouts as used here (contiguous k, wave32):
//   A[e]  : row = lane&15 (+16*ti), K = k0 + e + 16*hi
//   B[e]  : col = lane&15 (+16*tj), K = k0 + e + 16*hi
//   C/D[r]: row = r + 8*hi (+16*ti), col = lane&15 (+16*tj)
// ---------------------------------------------------------------------------
__global__ __launch_bounds__(256) void se_mlp_wmma_kernel(
    const float* __restrict__ sa,   // [32, 512]
    const float* __restrict__ sm,
    const float* __restrict__ sn,
    const float* __restrict__ w1,   // [MID=32, C=512] row-major
    const float* __restrict__ w2,   // [C=512, MID=32] row-major
    float* __restrict__ out)        // [32, 512] (-> [B,C,1,1])
{
    __shared__ float HS[BB][MID];   // Hsum = sum_v relu(V @ w1^T)

    const int tid  = threadIdx.x;
    const int wave = tid >> 5;
    const int lane = tid & 31;
    const int row  = lane & 15;     // A-row / B-col / D-col within a 16-tile
    const int hi   = lane >> 4;     // lane half selector

    // ---- GEMM1: 2x2 tiles of [32x32] = V[32x512] @ w1^T[512x32], 3 operands
    if (wave < 4) {
        const int ti = wave >> 1;   // batch tile (M)
        const int tj = wave & 1;    // MID tile  (N)
        v8f acc0 = {}, acc1 = {}, acc2 = {};

        for (int step = 0; step < CCH / 32; ++step) {   // 16 K-steps
            const int k0 = step << 5;

            v16h bm;                                    // shared B across the 3 Vs
#pragma unroll
            for (int e = 0; e < 16; ++e)
                bm[e] = (_Float16)w1[(tj * 16 + row) * CCH + k0 + e + 16 * hi];

            v16h a0, a1, a2;
#pragma unroll
            for (int e = 0; e < 16; ++e) {
                const int kk = k0 + e + 16 * hi;
                const int ai = (ti * 16 + row) * CCH + kk;
                a0[e] = (_Float16)sa[ai];
                a1[e] = (_Float16)sm[ai];
                a2[e] = (_Float16)sn[ai];
            }
            acc0 = __builtin_amdgcn_wmma_f32_16x16x32_f16(false, a0, false, bm,
                                                          (short)0, acc0, false, false);
            acc1 = __builtin_amdgcn_wmma_f32_16x16x32_f16(false, a1, false, bm,
                                                          (short)0, acc1, false, false);
            acc2 = __builtin_amdgcn_wmma_f32_16x16x32_f16(false, a2, false, bm,
                                                          (short)0, acc2, false, false);
        }

        // relu each operand's result, sum, park in LDS
#pragma unroll
        for (int r = 0; r < 8; ++r) {
            const float h = fmaxf(acc0[r], 0.0f) + fmaxf(acc1[r], 0.0f) + fmaxf(acc2[r], 0.0f);
            HS[ti * 16 + r + 8 * hi][tj * 16 + row] = h;
        }
    }
    __syncthreads();

    // ---- GEMM2: out[32x512] = HS[32x32] @ w2^T[32x512]; K = 32 -> 1 WMMA/tile
    for (int t = wave; t < 64; t += 8) {
        const int mi = t & 1;       // batch tile
        const int nj = t >> 1;      // channel tile (0..31)

        v16h a, bm;
#pragma unroll
        for (int e = 0; e < 16; ++e) {
            const int kk = e + 16 * hi;       // 0..31
            a[e]  = (_Float16)HS[mi * 16 + row][kk];
            bm[e] = (_Float16)w2[(nj * 16 + row) * MID + e + 16 * hi];
        }
        v8f acc = {};
        acc = __builtin_amdgcn_wmma_f32_16x16x32_f16(false, a, false, bm,
                                                     (short)0, acc, false, false);
#pragma unroll
        for (int r = 0; r < 8; ++r)
            out[(size_t)(mi * 16 + r + 8 * hi) * CCH + nj * 16 + row] = acc[r];
    }
}

// ---------------------------------------------------------------------------
extern "C" void kernel_launch(void* const* d_in, const int* in_sizes, int n_in,
                              void* d_out, int out_size, void* d_ws, size_t ws_size,
                              hipStream_t stream) {
    const float* x   = (const float*)d_in[0];   // [32,512,56,56]
    const float* dct = (const float*)d_in[1];   // [16,56,56]
    const float* w1  = (const float*)d_in[2];   // [32,512]
    const float* w2  = (const float*)d_in[3];   // [512,32]
    float* out = (float*)d_out;                 // [32,512,1,1]

    float* sa = (float*)d_ws;                   // 3 x 16384 floats of scratch
    float* sm = sa + BB * CCH;
    float* sn = sm + BB * CCH;

    // Kernel 1: 512 blocks x 256 threads; each block = 1 batch x 32 channels
    freq_pool_kernel<<<dim3(512), dim3(256), 0, stream>>>(x, dct, sa, sm, sn);

    // Kernel 2: single workgroup WMMA MLP
    se_mlp_wmma_kernel<<<dim3(1), dim3(256), 0, stream>>>(sa, sm, sn, w1, w2, out);
}
